// Qwen2AudioSAP2PoolingLayer_31516470018696
// MI455X (gfx1250) — compile-verified, matrix operand order checked by reading, not run on hardware
//
#include <hip/hip_runtime.h>
#include <hip/hip_bf16.h>

// Qwen2Audio SAP2 pooling layer for MI455X (gfx1250).
//
// Reference math collapses: softmax over q then sum over q == 1, so the
// attention weights are constant; the per-window softmax of a constant is
// exactly 1/window_size = 0.25. Result:
//   pooled[b,g,h]  = 0.25 * sum_{w<4} ctx[b, 4g+w, h]
//   pool_mask[b,g] = any(ctx_mask[b, 4g..4g+3])
// audio_hidden_states / enc_audio_mask are dead inputs.
//
// Implemented as a block-diagonal 16x16x64 matmul per output tile using
// v_wmma_f32_16x16x32_f16 (A = pooling weights {0, 0.25}, exact in f16;
// B = ctx rounded to f16; f32 accumulate). Fully HBM-bound (~42 MiB).

typedef __attribute__((ext_vector_type(16))) _Float16 v16h;
typedef __attribute__((ext_vector_type(8)))  float    v8f;

#define B_   2
#define CTX_ 4096
#define H_   1024
#define W_   4
#define G_   (CTX_ / W_)   // 1024 pooled groups

// One wave computes pooled[b, g0:g0+16, h0:h0+16].
// Workgroup = 8 waves tiling 128 hidden channels.
__global__ __launch_bounds__(256) void sap2_pool_wmma(const float* __restrict__ ctx,
                                                      float* __restrict__ out) {
  const int wave = threadIdx.x >> 5;   // wave32
  const int lane = threadIdx.x & 31;
  const int m    = lane & 15;          // A row / B column / D column index
  const int half = lane >> 4;          // lane-half selector per ISA VGPR layout

  const int b  = blockIdx.z;
  const int g0 = blockIdx.y * 16;
  const int h0 = (blockIdx.x * 8 + wave) * 16;
  const int r0 = g0 * W_;              // first ctx row of this tile (64 rows total)

  const float* cbase = ctx + (size_t)b * CTX_ * H_;
  const int col = h0 + m;

  // CDNA5 speculative prefetch of this tile's first cachelines.
  __builtin_prefetch(cbase + (size_t)r0 * H_ + col, 0, 1);

  // Operand build per ISA 16-bit 16x32 A layout (wave32):
  //   VGPR v in 0..7, lane-half h: packed K pair = {k0, k0+1},
  //   k0 = 2*(v&3) + 8*half + 16*(v>>2).
  // A#0 covers K=0..31 (ctx rows r0+K), A#1 covers K=32..63.
  // Pooling matrix: A[m][k] = 0.25 iff k/4 == m (block diagonal), exact in f16.
  v16h a0 = {}, a1 = {}, bm0, bm1;
  #pragma unroll
  for (int v = 0; v < 8; ++v) {
    const int k0 = 2 * (v & 3) + 8 * half + 16 * (v >> 2);
    // k0 even => k0 and k0+1 share the same k/4 window.
    const _Float16 w0 = ((k0 >> 2)        == m) ? (_Float16)0.25f : (_Float16)0.0f;
    const _Float16 w1 = (((k0 + 32) >> 2) == m) ? (_Float16)0.25f : (_Float16)0.0f;
    a0[2 * v]     = w0;
    a0[2 * v + 1] = w0;
    a1[2 * v]     = w1;
    a1[2 * v + 1] = w1;

    // B (32x16 per step): lane's column = col, rows = K (mirrored striping).
    bm0[2 * v]     = (_Float16)cbase[(size_t)(r0 + k0)      * H_ + col];
    bm0[2 * v + 1] = (_Float16)cbase[(size_t)(r0 + k0 + 1)  * H_ + col];
    bm1[2 * v]     = (_Float16)cbase[(size_t)(r0 + k0 + 32) * H_ + col];
    bm1[2 * v + 1] = (_Float16)cbase[(size_t)(r0 + k0 + 33) * H_ + col];
  }

  // D = A x B + C, f32 accumulate, K=64 via two chained WMMAs.
  v8f c = {};
  c = __builtin_amdgcn_wmma_f32_16x16x32_f16(false, a0, false, bm0,
                                             (short)0, c, false, false);
  c = __builtin_amdgcn_wmma_f32_16x16x32_f16(false, a1, false, bm1,
                                             (short)0, c, false, false);

  // Scatter D per 32-bit C/D layout: VGPR r -> row M = r + 8*half, col N = m.
  float* obase = out + ((size_t)b * G_ + (size_t)g0) * H_;
  #pragma unroll
  for (int r = 0; r < 8; ++r) {
    obase[(size_t)(r + 8 * half) * H_ + col] = c[r];
  }
}

// pool_mask[b,g] = (sum of 4 window mask entries > 0) ? 1 : 0, stored as float
// in the concatenated output tail.
__global__ __launch_bounds__(256) void sap2_pool_mask(const int* __restrict__ cmask,
                                                      float* __restrict__ outm) {
  const int idx = blockIdx.x * blockDim.x + threadIdx.x;
  if (idx >= B_ * G_) return;
  const int b = idx / G_;
  const int g = idx - b * G_;
  const int* p = cmask + (size_t)b * CTX_ + (size_t)g * W_;
  const int s = p[0] + p[1] + p[2] + p[3];
  outm[idx] = (s > 0) ? 1.0f : 0.0f;
}

extern "C" void kernel_launch(void* const* d_in, const int* in_sizes, int n_in,
                              void* d_out, int out_size, void* d_ws, size_t ws_size,
                              hipStream_t stream) {
  (void)in_sizes; (void)n_in; (void)out_size; (void)d_ws; (void)ws_size;

  // d_in[0]: audio_hidden_states (dead), d_in[1]: contxt_hidden_states f32,
  // d_in[2]: enc_audio_mask (dead),      d_in[3]: enc_contxt_mask i32,
  // d_in[4]: window_size (=4, static).
  const float* ctx   = (const float*)d_in[1];
  const int*   cmask = (const int*)d_in[3];
  float*       out   = (float*)d_out;

  // 8192 tiles of 16x16: grid.x tiles hidden (8 wg x 8 waves x 16 = 1024),
  // grid.y tiles groups (64 x 16 = 1024), grid.z = batch.
  dim3 grid(H_ / (8 * 16), G_ / 16, B_);
  sap2_pool_wmma<<<grid, dim3(256), 0, stream>>>(ctx, out);

  const int nmask = B_ * G_;
  sap2_pool_mask<<<(nmask + 255) / 256, 256, 0, stream>>>(
      cmask, out + (size_t)B_ * G_ * H_);
}
